// Model_14353780703773
// MI455X (gfx1250) — matrix operand order repaired and checked by reading, not compiled, not run on hardware
//
#include <hip/hip_runtime.h>
#include <stdint.h>

#define SLOPE 0.2f
#define B_    32
#define LL_   512
#define D_    64
#define H_    64
#define LEN_  510                 // LL - LAGS
#define NROWS_ (B_*LEN_)          // 16320
#define NT_    (NROWS_/16)        // 1020 row-tiles of 16
#define WPB_   4                  // waves (n-tiles) per block
#define NG_    (NT_/WPB_)         // 255 n-groups

typedef __attribute__((ext_vector_type(16))) __bf16 v16bf;
typedef __attribute__((ext_vector_type(8)))  float  v8f;
typedef __attribute__((ext_vector_type(4)))  unsigned int u32x4;
typedef __attribute__((ext_vector_type(8)))  int i32x8;
typedef __attribute__((ext_vector_type(4)))  int i32x4;

#if defined(__has_builtin)
#if __has_builtin(__builtin_amdgcn_tensor_load_to_lds)
#define HAVE_TDM 1
#endif
#endif

union Frag16 { v16bf v; uint4 u[2]; };

__device__ __forceinline__ unsigned short f2bf(float f) {
    union { float f; unsigned int u; } x; x.f = f;
    unsigned int u = x.u;
    unsigned int r = u + 0x7FFFu + ((u >> 16) & 1u);   // round-to-nearest-even
    return (unsigned short)(r >> 16);
}

// ---- prep 1: x (f32) -> bf16 ----
__global__ void cvt_x_bf16(const float* __restrict__ x, unsigned short* __restrict__ xbf, int n) {
    int i = blockIdx.x * blockDim.x + threadIdx.x;
    if (i < n) xbf[i] = f2bf(x[i]);
}

// ---- prep 2: pack W1[:, :, 0:128] into WMMA B-fragment layout (bf16) ----
// frag index = ((d*4 + ht)*4 + ks); each frag = 32 lanes x 16 bf16 (32B/lane).
__global__ void pack_w1(const float* __restrict__ W1, unsigned short* __restrict__ bp) {
    int tid = blockIdx.x * blockDim.x + threadIdx.x;   // ((d*4+ht)*4+ks)*32 + lane
    if (tid >= 64 * 4 * 4 * 32) return;
    int lane = tid & 31;
    int ks   = (tid >> 5) & 3;
    int ht   = (tid >> 7) & 3;
    int d    = tid >> 9;
    int h    = ht * 16 + (lane & 15);
    int kb   = ks * 32 + (lane >> 4) * 16;
    const float* src = W1 + (d * H_ + h) * 129 + kb;   // 16 consecutive f32
    unsigned short* dst = bp + (size_t)tid * 16;
#pragma unroll
    for (int i = 0; i < 16; ++i) dst[i] = f2bf(src[i]);
}

// ---- main: 4 waves/block, one d per block, B slab staged to LDS via TDM ----
__global__ __launch_bounds__(32 * WPB_) void gemm_block(
    const unsigned short* __restrict__ xbf,
    const unsigned short* __restrict__ bpack,
    const float* __restrict__ x,
    const float* __restrict__ W1,
    const float* __restrict__ b1,
    const float* __restrict__ W2,
    const float* __restrict__ b2,
    float* __restrict__ out_res,   // (B, LEN, D) flat: n*64 + d
    float* __restrict__ Jws)       // (NROWS, D)
{
    __shared__ unsigned short Bslab[4 * 4 * 32 * 16];  // 8192 bf16 = 16 KB (one d)
    __shared__ float tiles[WPB_][16 * 64];             // per-wave C staging
    __shared__ float w2s[64], wls[64], b1s[64];

    int d    = blockIdx.x & 63;
    int ng   = blockIdx.x >> 6;
    int tid  = threadIdx.x;
    int wv   = tid >> 5;           // wave in block
    int lane = tid & 31;
    int half = lane >> 4;
    int m0   = lane & 15;

    // ---- stage B slab for this d into LDS ----
#if HAVE_TDM
    if (wv == 0) {
        unsigned long long ga = (unsigned long long)(uintptr_t)(bpack + (size_t)d * 8192);
        unsigned int lds_addr = (unsigned int)(uintptr_t)(&Bslab[0]);   // addr[31:0] = LDS offset
        // D# group 0: count=1, lds_addr, global_addr[56:0], type=2
        u32x4 g0 = { 1u,
                     lds_addr,
                     (unsigned int)ga,
                     (unsigned int)((ga >> 32) & 0x01FFFFFFu) | (2u << 30) };
        // D# group 1: data_size=2B (code 1); tensor_dim0=8192; tensor_dim1=1;
        //             tile_dim0=8192; tile_dim1=1; tensor_dim0_stride=8192
        i32x8 g1 = { (int)(1u << 16),                  // data_size @ [17:16]
                     (int)((8192u & 0xFFFFu) << 16),   // tensor_dim0 lo16 @ bits[63:48]
                     (int)((8192u >> 16) | (1u << 16)),// dim0 hi16 | tensor_dim1 lo16 @ [95:80]
                     (int)(8192u << 16),               // dim1 hi16=0 | tile_dim0 @ [127:112]
                     1,                                // tile_dim1=1 @ [143:128], tile_dim2=0
                     8192,                             // tensor_dim0_stride lo32 @ [191:160]
                     0, 0 };
        i32x4 gz4 = { 0, 0, 0, 0 };                    // groups 2/3 unused (<=2D)
        i32x8 gz8 = { 0, 0, 0, 0, 0, 0, 0, 0 };
        __builtin_amdgcn_tensor_load_to_lds(g0, g1, gz4, gz4, gz8, 0);
        __builtin_amdgcn_s_wait_tensorcnt(0);
    }
#else
    {   // fallback: cooperative copy, 128 threads x 128B
        const uint4* src = (const uint4*)(bpack + (size_t)d * 8192);
        uint4* dst = (uint4*)&Bslab[0];
        for (int i = tid; i < 1024; i += 32 * WPB_) dst[i] = src[i];
    }
#endif
    // per-d params (f32)
    if (tid < 64) {
        w2s[tid] = W2[d * 64 + tid];
        wls[tid] = W1[(d * 64 + tid) * 129 + 128];
        b1s[tid] = b1[d * 64 + tid];
    }
    __syncthreads();

    // ---- GEMM: this wave's 16-row tile ----
    int nt = ng * WPB_ + wv;
    int n_base = nt * 16;
    int n_row = n_base + m0;
    int bb = n_row / LEN_;
    int tt = n_row - bb * LEN_;
    const unsigned short* rowp = xbf + ((size_t)(bb * LL_ + tt)) * D_;

    Frag16 a[4];
#pragma unroll
    for (int ks = 0; ks < 4; ++ks) {
        int k0 = ks * 32 + half * 16;
        a[ks].u[0] = *(const uint4*)(rowp + k0);
        a[ks].u[1] = *(const uint4*)(rowp + k0 + 8);
    }

    v8f acc[4] = {};
#pragma unroll
    for (int ht = 0; ht < 4; ++ht) {
#pragma unroll
        for (int ks = 0; ks < 4; ++ks) {
            Frag16 bf;
            const uint4* bp4 = (const uint4*)&Bslab[((ht * 4 + ks) * 32 + lane) * 16];
            bf.u[0] = bp4[0];
            bf.u[1] = bp4[1];
            acc[ht] = __builtin_amdgcn_wmma_f32_16x16x32_bf16(
                false, a[ks].v, false, bf.v, (short)0, acc[ht], false, false);
        }
    }

    // stage C tile: element (m = v + 8*half, h = ht*16 + m0)
#pragma unroll
    for (int ht = 0; ht < 4; ++ht) {
        int h = ht * 16 + m0;
#pragma unroll
        for (int v = 0; v < 8; ++v)
            tiles[wv][(v + half * 8) * 64 + h] = acc[ht][v];
    }
    __syncthreads();

    // ---- f32 epilogue, full wave: lane handles row m0, h-range [half*32, half*32+32) ----
    {
        int n  = n_base + m0;
        int b_ = n / LEN_;
        int t_ = n - b_ * LEN_;
        float xt = x[((size_t)(b_ * LL_ + t_ + 2)) * D_ + d];
        float racc = 0.f, jacc = 0.f;
        int h0 = half * 32;
#pragma unroll 8
        for (int i = 0; i < 32; ++i) {
            int h = h0 + i;
            float pre = tiles[wv][m0 * 64 + h] + xt * wls[h] + b1s[h];
            float act, da;
            if (pre >= 0.f) { act = pre;         da = 1.f;   }
            else            { act = SLOPE * pre; da = SLOPE; }
            racc += act * w2s[h];
            jacc += da * w2s[h] * wls[h];
        }
        racc += __shfl_xor(racc, 16, 32);
        jacc += __shfl_xor(jacc, 16, 32);
        if (lane < 16) {
            out_res[(size_t)n * D_ + d] = racc + b2[d];
            Jws[(size_t)n * D_ + d]     = jacc;
        }
    }
}

// ---- final: sum_d log|J[n,d]| ----
__global__ void reduce_logdet(const float* __restrict__ Jws, float* __restrict__ outld) {
    int n = blockIdx.x * blockDim.x + threadIdx.x;
    if (n >= NROWS_) return;
    const float* p = Jws + (size_t)n * 64;
    float s = 0.f;
#pragma unroll 8
    for (int d = 0; d < 64; ++d) s += logf(fabsf(p[d]));
    outld[n] = s;
}

extern "C" void kernel_launch(void* const* d_in, const int* in_sizes, int n_in,
                              void* d_out, int out_size, void* d_ws, size_t ws_size,
                              hipStream_t stream) {
    const float* x  = (const float*)d_in[0];
    const float* W1 = (const float*)d_in[1];
    const float* b1 = (const float*)d_in[2];
    const float* W2 = (const float*)d_in[3];
    const float* b2 = (const float*)d_in[4];

    float* out_res = (float*)d_out;                 // 32*510*64 floats
    float* out_ld  = out_res + (size_t)NROWS_ * D_; // 16320 floats

    char* ws = (char*)d_ws;
    unsigned short* xbf  = (unsigned short*)ws;                    // 2 MB
    unsigned short* bpak = (unsigned short*)(ws + 2097152);        // 1 MB
    float*          Jws  = (float*)(ws + 3145728);                 // ~4.2 MB

    int nx = B_ * LL_ * D_;
    cvt_x_bf16<<<(nx + 255) / 256, 256, 0, stream>>>(x, xbf, nx);
    pack_w1<<<(16384 + 255) / 256, 256, 0, stream>>>(W1, bpak);
    gemm_block<<<NG_ * 64, 32 * WPB_, 0, stream>>>(xbf, bpak, x, W1, b1, W2, b2, out_res, Jws);
    reduce_logdet<<<(NROWS_ + 255) / 256, 256, 0, stream>>>(Jws, out_ld);
}
